// Decoder_46325517255231
// MI455X (gfx1250) — compile-verified, matrix operand order checked by reading, not run on hardware
//
#include <hip/hip_runtime.h>
#include <hip/hip_bf16.h>
#include <math.h>

// Shapes
#define Bb 128
#define Tt 256
#define Ss 256
#define Hh 512
#define Ee 256

typedef __attribute__((ext_vector_type(16))) __bf16 v16bf;
typedef __attribute__((ext_vector_type(8)))  __bf16 v8bf;
typedef __attribute__((ext_vector_type(8)))  float  v8f;
typedef int v4i __attribute__((vector_size(16)));   // matches async-LDS builtin param

// ---- gfx1250 async global->LDS staging path -----------------------------
#if defined(__has_builtin)
#if __has_builtin(__builtin_amdgcn_global_load_async_to_lds_b128)
#define HAVE_ASYNC_LDS 1
#endif
#if __has_builtin(__builtin_amdgcn_s_wait_asynccnt)
#define HAVE_WAIT_ASYNC 1
#endif
#endif
#ifndef HAVE_ASYNC_LDS
#define HAVE_ASYNC_LDS 0
#endif
#ifndef HAVE_WAIT_ASYNC
#define HAVE_WAIT_ASYNC 0
#endif

// Copy 16 bytes global -> LDS (async DMA on gfx1250 when available).
__device__ __forceinline__ void stage16(const __bf16* g, __bf16* l) {
#if HAVE_ASYNC_LDS
    __builtin_amdgcn_global_load_async_to_lds_b128(
        (__attribute__((address_space(1))) v4i*)g,
        (__attribute__((address_space(3))) v4i*)l, 0, 0);
#else
    *(v8bf*)l = *(const v8bf*)g;
#endif
}
__device__ __forceinline__ void stage_fence() {
#if HAVE_ASYNC_LDS
#if HAVE_WAIT_ASYNC
    __builtin_amdgcn_s_wait_asynccnt(0);
#else
    asm volatile("s_wait_asynccnt 0x0" ::: "memory");
#endif
#endif
}

// fp32 -> bf16 round-to-nearest-even (used only OFF the GEMM hot paths)
__device__ __forceinline__ __bf16 f2bf(float f) {
    unsigned u = __builtin_bit_cast(unsigned, f);
    unsigned short s = (unsigned short)((u + 0x7FFFu + ((u >> 16) & 1u)) >> 16);
    return __builtin_bit_cast(__bf16, s);
}

// 16-half fragment: elems 0..7 = p[0..7], elems 8..15 = p[16..23]
// (p already includes the lane's kh = (lane>>4)*8 offset in the 32-wide K tile).
__device__ __forceinline__ v16bf frag(const __bf16* p) {
    v8bf lo = *(const v8bf*)p;
    v8bf hi = *(const v8bf*)(p + 16);
    return __builtin_shufflevector(lo, hi, 0, 1, 2, 3, 4, 5, 6, 7,
                                   8, 9, 10, 11, 12, 13, 14, 15);
}

__device__ __forceinline__ v8f wmma_bf16(v16bf a, v16bf b, v8f c) {
    return __builtin_amdgcn_wmma_f32_16x16x32_bf16(false, a, false, b, (short)0, c,
                                                   false, false);
}

__device__ __forceinline__ float sigf(float x) { return 1.0f / (1.0f + expf(-x)); }

// ---------------------------------------------------------------- fp32 -> bf16 copy
__global__ void cvt_bf16_kernel(const float* __restrict__ src,
                                __bf16* __restrict__ dst, int n8) {
    int i = blockIdx.x * blockDim.x + threadIdx.x;
    if (i >= n8) return;
    const float* s = src + (size_t)i * 8;
    v8bf o;
#pragma unroll
    for (int k = 0; k < 8; ++k) o[k] = f2bf(s[k]);
    *(v8bf*)(dst + (size_t)i * 8) = o;
}

// ---------------------------------------------------------------- bridge
__global__ void bridge_kernel(const float* __restrict__ eh, const float* __restrict__ ec,
                              const float* __restrict__ hW, const float* __restrict__ hb,
                              const float* __restrict__ cW, const float* __restrict__ cb,
                              float* __restrict__ h0, float* __restrict__ c0) {
    int idx = blockIdx.x * blockDim.x + threadIdx.x;
    int which = idx >> 16;
    int i = idx & 65535;
    int b = i >> 9, n = i & 511;
    const float* x = (which ? ec : eh) + (size_t)b * (2 * Hh);
    const float* w = (which ? cW : hW) + (size_t)n * (2 * Hh);
    float acc = which ? cb[n] : hb[n];
    for (int k = 0; k < 2 * Hh; ++k) acc += x[k] * w[k];
    (which ? c0 : h0)[i] = tanhf(acc);
}

// ---------------------------------------------------------------- qproj + hterm
__global__ void qh_kernel(const float* __restrict__ h0, const float* __restrict__ qW,
                          const float* __restrict__ Whh, const float* __restrict__ bih,
                          const float* __restrict__ bhh, float* __restrict__ qproj,
                          float* __restrict__ hterm) {
    int idx = blockIdx.x * blockDim.x + threadIdx.x;
    if (idx < Bb * Hh) {
        int b = idx >> 9, n = idx & 511;
        const float* x = h0 + (size_t)b * Hh;
        const float* w = qW + (size_t)n * Hh;
        float acc = 0.f;
        for (int k = 0; k < Hh; ++k) acc += x[k] * w[k];
        qproj[idx] = acc;
    } else {
        int i = idx - Bb * Hh;
        int b = i >> 11, n = i & 2047;
        const float* x = h0 + (size_t)b * Hh;
        const float* w = Whh + (size_t)n * Hh;
        float acc = bih[n] + bhh[n];
        for (int k = 0; k < Hh; ++k) acc += x[k] * w[k];
        hterm[i] = acc;
    }
}

// ---------------------------------------------------------------- attention scores
// K-outer / N-inner: each wave holds acc[8 ntiles][2 row tiles] (128 VGPRs),
// A fragments loaded once per K-step from LDS, B streamed from L2.
#define ATT_STRIDE (1024 + 8)
__global__ __launch_bounds__(128, 2)
void attn_scores_kernel(const __bf16* __restrict__ enc_bf,
                        const __bf16* __restrict__ keyW_bf,
                        const float* __restrict__ qproj, const float* __restrict__ eW,
                        float* __restrict__ scores) {
    const int rowbase = blockIdx.x * 32;               // 1024 blocks
    const int b = rowbase >> 8;
    const int wave = threadIdx.x >> 5;
    const int lane = threadIdx.x & 31;
    const int col = lane & 15;
    const int kh = (lane >> 4) * 8;

    __shared__ __align__(16) __bf16 As[32 * ATT_STRIDE];
    __shared__ float rowsum[32];
    if (threadIdx.x < 32) rowsum[threadIdx.x] = 0.f;
    for (int c = threadIdx.x; c < 32 * 128; c += 128) {   // 8-half chunks
        int row = c >> 7;
        int k = (c & 127) * 8;
        stage16(enc_bf + (size_t)(rowbase + row) * 1024 + k,
                &As[row * ATT_STRIDE + k]);
    }
    stage_fence();
    __syncthreads();

    // wave owns 8 contiguous N-tiles: n = wave*128 + i*16 + col
    const __bf16* B0 = keyW_bf + (size_t)(wave * 128 + col) * 1024;
    v8f acc0[8] = {}, acc1[8] = {};
#pragma unroll 1
    for (int ks = 0; ks < 32; ++ks) {
        const int kb = ks * 32 + kh;
        v16bf a0 = frag(&As[col * ATT_STRIDE + kb]);
        v16bf a1 = frag(&As[(16 + col) * ATT_STRIDE + kb]);
#pragma unroll
        for (int i = 0; i < 8; ++i) {
            v16bf w = frag(B0 + (size_t)i * 16 * 1024 + kb);
            acc0[i] = wmma_bf16(a0, w, acc0[i]);
            acc1[i] = wmma_bf16(a1, w, acc1[i]);
        }
    }
#pragma unroll
    for (int i = 0; i < 8; ++i) {
        const int n = wave * 128 + i * 16 + col;
        const float q = qproj[b * Hh + n];
        const float e = eW[n];
#pragma unroll
        for (int v = 0; v < 8; ++v) {
            const int m = v + ((lane >> 4) << 3);
            atomicAdd(&rowsum[m],      tanhf(acc0[i][v] + q) * e);
            atomicAdd(&rowsum[16 + m], tanhf(acc1[i][v] + q) * e);
        }
    }
    __syncthreads();
    if (threadIdx.x < 32) scores[rowbase + threadIdx.x] = rowsum[threadIdx.x];
}

// ---------------------------------------------------------------- softmax
__global__ void softmax_kernel(const float* __restrict__ scores,
                               const int* __restrict__ src_pos,
                               float* __restrict__ alphas) {
    int b = blockIdx.x, s = threadIdx.x;
    __shared__ float red[Ss];
    float v = scores[b * Ss + s];
    if (src_pos[b * Ss + s] == 0) v = -1.0e9f;
    red[s] = v; __syncthreads();
    for (int off = 128; off > 0; off >>= 1) {
        if (s < off) red[s] = fmaxf(red[s], red[s + off]);
        __syncthreads();
    }
    float mx = red[0]; __syncthreads();
    float e = expf(v - mx);
    red[s] = e; __syncthreads();
    for (int off = 128; off > 0; off >>= 1) {
        if (s < off) red[s] += red[s + off];
        __syncthreads();
    }
    alphas[b * Ss + s] = e / red[0];
}

// ---------------------------------------------------------------- context (-> bf16)
__global__ void context_kernel(const float* __restrict__ alphas,
                               const float* __restrict__ enc,
                               __bf16* __restrict__ ctx_bf) {
    int b = blockIdx.x;
    __shared__ float al[Ss];
    al[threadIdx.x] = alphas[b * Ss + threadIdx.x];
    __syncthreads();
    for (int d = threadIdx.x; d < 2 * Hh; d += 256) {
        const float* e = enc + (size_t)b * Ss * (2 * Hh) + d;
        float acc = 0.f;
        for (int s = 0; s < Ss; ++s) acc += al[s] * e[(size_t)s * (2 * Hh)];
        ctx_bf[b * (2 * Hh) + d] = f2bf(acc);
    }
}

// ---------------------------------------------------------------- gates + LSTM
// K-outer / gate-inner: acc[2 row tiles][4 gates] (64 VGPRs).
#define G_STRIDE (1280 + 8)
__global__ __launch_bounds__(128, 2)
void gates_lstm_kernel(const int* __restrict__ tgt_seq, const __bf16* __restrict__ emb_bf,
                       const __bf16* __restrict__ ctx_bf, const __bf16* __restrict__ Wih_bf,
                       const float* __restrict__ hterm, const float* __restrict__ c0,
                       __bf16* __restrict__ hbuf_bf) {
    const int nt = (blockIdx.x << 2) | (threadIdx.x >> 5);  // 0..31
    const int rowbase = blockIdx.y * 32;                    // 1024 row blocks
    const int b = rowbase >> 8;
    const int lane = threadIdx.x & 31;
    const int col = lane & 15;
    const int kh = (lane >> 4) * 8;

    __shared__ __align__(16) __bf16 As[32 * G_STRIDE];
    __shared__ int tgs[32];
    if (threadIdx.x < 32)
        tgs[threadIdx.x] = tgt_seq[b * Tt + ((rowbase + threadIdx.x) & 255)];
    __syncthreads();
    for (int c = threadIdx.x; c < 32 * 160; c += 128) {     // 8-half chunks
        int row = c / 160;
        int k = (c % 160) * 8;
        const __bf16* src = (k < Ee) ? emb_bf + (size_t)tgs[row] * Ee + k
                                     : ctx_bf + (size_t)b * (2 * Hh) + (k - Ee);
        stage16(src, &As[row * G_STRIDE + k]);
    }
    stage_fence();
    __syncthreads();

    const int j = nt * 16 + col;
    const __bf16* Wb = Wih_bf + (size_t)j * 1280;           // gate g at +g*512*1280

    v8f acc[2][4] = {};
#pragma unroll 1
    for (int ks = 0; ks < 40; ++ks) {                       // K = 1280
        const int kb = ks * 32 + kh;
        v16bf a0 = frag(&As[col * G_STRIDE + kb]);
        v16bf a1 = frag(&As[(16 + col) * G_STRIDE + kb]);
#pragma unroll
        for (int g = 0; g < 4; ++g) {
            v16bf w = frag(Wb + (size_t)g * Hh * 1280 + kb);
            acc[0][g] = wmma_bf16(a0, w, acc[0][g]);
            acc[1][g] = wmma_bf16(a1, w, acc[1][g]);
        }
    }
    const float c0v = c0[b * Hh + j];
    const float hti = hterm[b * (4 * Hh) + 0 * Hh + j];
    const float htf = hterm[b * (4 * Hh) + 1 * Hh + j];
    const float htg = hterm[b * (4 * Hh) + 2 * Hh + j];
    const float hto = hterm[b * (4 * Hh) + 3 * Hh + j];
#pragma unroll
    for (int half = 0; half < 2; ++half) {
#pragma unroll
        for (int v = 0; v < 8; ++v) {
            const int m = v + ((lane >> 4) << 3);
            const int row = rowbase + half * 16 + m;
            float i_ = acc[half][0][v] + hti;
            float f_ = acc[half][1][v] + htf;
            float g_ = acc[half][2][v] + htg;
            float o_ = acc[half][3][v] + hto;
            float c = sigf(f_) * c0v + sigf(i_) * tanhf(g_);
            hbuf_bf[(size_t)row * Hh + j] = f2bf(sigf(o_) * tanhf(c));
        }
    }
}

// ---------------------------------------------------------------- output GEMM
// K-outer / N-inner: each wave holds acc[4 ntiles][2 row tiles] (64 VGPRs).
#define O_STRIDE (1792 + 8)
__global__ __launch_bounds__(128, 2)
void out_gemm_kernel(const int* __restrict__ tgt_seq, const __bf16* __restrict__ emb_bf,
                     const __bf16* __restrict__ hbuf_bf, const __bf16* __restrict__ ctx_bf,
                     const __bf16* __restrict__ preW_bf, float* __restrict__ out) {
    const int ntbase = blockIdx.x * 16 + (threadIdx.x >> 5) * 4;  // 0..28 step 4
    const int rowbase = blockIdx.y * 32;                          // 1024 row blocks
    const int b = rowbase >> 8;
    const int lane = threadIdx.x & 31;
    const int col = lane & 15;
    const int kh = (lane >> 4) * 8;

    __shared__ __align__(16) __bf16 As[32 * O_STRIDE];
    __shared__ int tgs[32];
    if (threadIdx.x < 32)
        tgs[threadIdx.x] = tgt_seq[b * Tt + ((rowbase + threadIdx.x) & 255)];
    __syncthreads();
    for (int c = threadIdx.x; c < 32 * 224; c += 128) {     // 8-half chunks
        int row = c / 224;
        int k = (c % 224) * 8;
        int r = rowbase + row;
        const __bf16* src;
        if (k < Ee)            src = emb_bf + (size_t)tgs[row] * Ee + k;
        else if (k < Ee + Hh)  src = hbuf_bf + (size_t)r * Hh + (k - Ee);
        else                   src = ctx_bf + (size_t)b * (2 * Hh) + (k - Ee - Hh);
        stage16(src, &As[row * O_STRIDE + k]);
    }
    stage_fence();
    __syncthreads();

    const __bf16* B0 = preW_bf + (size_t)(ntbase * 16 + col) * 1792;
    v8f acc0[4] = {}, acc1[4] = {};
#pragma unroll 1
    for (int ks = 0; ks < 56; ++ks) {                       // K = 1792
        const int kb = ks * 32 + kh;
        v16bf a0 = frag(&As[col * O_STRIDE + kb]);
        v16bf a1 = frag(&As[(16 + col) * O_STRIDE + kb]);
#pragma unroll
        for (int i = 0; i < 4; ++i) {
            v16bf w = frag(B0 + (size_t)i * 16 * 1792 + kb);
            acc0[i] = wmma_bf16(a0, w, acc0[i]);
            acc1[i] = wmma_bf16(a1, w, acc1[i]);
        }
    }
#pragma unroll
    for (int i = 0; i < 4; ++i) {
        const int n = (ntbase + i) * 16 + col;
#pragma unroll
        for (int v = 0; v < 8; ++v) {
            const int m = v + ((lane >> 4) << 3);
            out[(size_t)(rowbase + m) * Hh + n]      = acc0[i][v];
            out[(size_t)(rowbase + 16 + m) * Hh + n] = acc1[i][v];
        }
    }
}

// ---------------------------------------------------------------- launcher
extern "C" void kernel_launch(void* const* d_in, const int* in_sizes, int n_in,
                              void* d_out, int out_size, void* d_ws, size_t ws_size,
                              hipStream_t stream) {
    const int*   tgt_seq = (const int*)  d_in[0];
    const float* enc     = (const float*)d_in[1];
    const float* eh      = (const float*)d_in[2];
    const float* ec      = (const float*)d_in[3];
    const int*   src_pos = (const int*)  d_in[4];
    const float* emb     = (const float*)d_in[5];
    const float* keyW    = (const float*)d_in[6];
    const float* qW      = (const float*)d_in[7];
    const float* eW      = (const float*)d_in[8];
    const float* Wih     = (const float*)d_in[9];
    const float* Whh     = (const float*)d_in[10];
    const float* bih     = (const float*)d_in[11];
    const float* bhh     = (const float*)d_in[12];
    const float* bhW     = (const float*)d_in[13];
    const float* bhb     = (const float*)d_in[14];
    const float* bcW     = (const float*)d_in[15];
    const float* bcb     = (const float*)d_in[16];
    const float* preW    = (const float*)d_in[17];
    float* out = (float*)d_out;

    // ---- workspace layout (16B aligned throughout), ~117 MB total
    float* ws     = (float*)d_ws;
    float* h0     = ws;                     // 128*512
    float* c0     = h0 + Bb * Hh;           // 128*512
    float* qproj  = c0 + Bb * Hh;           // 128*512
    float* hterm  = qproj + Bb * Hh;        // 128*2048
    float* scores = hterm + Bb * 4 * Hh;    // 128*256
    float* alphas = scores + Bb * Ss;       // 128*256
    __bf16* bfbase  = (__bf16*)(alphas + Bb * Ss);
    __bf16* enc_bf  = bfbase;                                    // 128*256*1024
    __bf16* keyW_bf = enc_bf + (size_t)Bb * Ss * 2 * Hh;         // 512*1024
    __bf16* Wih_bf  = keyW_bf + (size_t)Hh * 2 * Hh;             // 2048*1280
    __bf16* preW_bf = Wih_bf + (size_t)4 * Hh * (Ee + 2 * Hh);   // 512*1792
    __bf16* emb_bf  = preW_bf + (size_t)Hh * (Ee + Hh + 2 * Hh); // 10000*256
    __bf16* ctx_bf  = emb_bf + (size_t)10000 * Ee;               // 128*1024
    __bf16* hbuf_bf = ctx_bf + (size_t)Bb * 2 * Hh;              // 128*256*512

    auto cvt = [&](const float* s, __bf16* d, size_t n) {
        int n8 = (int)(n / 8);
        cvt_bf16_kernel<<<(n8 + 255) / 256, 256, 0, stream>>>(s, d, n8);
    };
    cvt(enc,  enc_bf,  (size_t)Bb * Ss * 2 * Hh);
    cvt(keyW, keyW_bf, (size_t)Hh * 2 * Hh);
    cvt(Wih,  Wih_bf,  (size_t)4 * Hh * (Ee + 2 * Hh));
    cvt(preW, preW_bf, (size_t)Hh * (Ee + Hh + 2 * Hh));
    cvt(emb,  emb_bf,  (size_t)10000 * Ee);

    bridge_kernel<<<512, 256, 0, stream>>>(eh, ec, bhW, bhb, bcW, bcb, h0, c0);
    qh_kernel<<<1280, 256, 0, stream>>>(h0, qW, Whh, bih, bhh, qproj, hterm);
    attn_scores_kernel<<<1024, 128, 0, stream>>>(enc_bf, keyW_bf, qproj, eW, scores);
    softmax_kernel<<<Bb, Ss, 0, stream>>>(scores, src_pos, alphas);
    context_kernel<<<Bb, 256, 0, stream>>>(alphas, enc, ctx_bf);
    gates_lstm_kernel<<<dim3(8, 1024), 128, 0, stream>>>(tgt_seq, emb_bf, ctx_bf,
                                                         Wih_bf, hterm, c0, hbuf_bf);
    out_gemm_kernel<<<dim3(2, 1024), 128, 0, stream>>>(tgt_seq, emb_bf, hbuf_bf,
                                                       ctx_bf, preW_bf, out);
}
